// ClusteringLayer_7215545057821
// MI455X (gfx1250) — compile-verified, matrix operand order checked by reading, not run on hardware
//
#include <hip/hip_runtime.h>

#define NFEAT 128
#define NCLUS 64
#define PFDIST 3   // prefetch distance in grid-stride iterations (covers HBM latency)

typedef __bf16 v16bf __attribute__((ext_vector_type(16)));
typedef float  v8f   __attribute__((ext_vector_type(8)));

__device__ __forceinline__ unsigned fkey(float f) {
    // map float -> uint so unsigned compare == float compare (ascending)
    unsigned u = __float_as_uint(f);
    return (u & 0x80000000u) ? ~u : (u | 0x80000000u);
}

__device__ __forceinline__ float ssq4(float4 a) {
    return a.x * a.x + a.y * a.y + a.z * a.z + a.w * a.w;
}

__device__ __forceinline__ v16bf pack16(float4 a, float4 b, float4 c, float4 d, float scale) {
    v16bf v;
    v[0]  = (__bf16)(a.x * scale); v[1]  = (__bf16)(a.y * scale);
    v[2]  = (__bf16)(a.z * scale); v[3]  = (__bf16)(a.w * scale);
    v[4]  = (__bf16)(b.x * scale); v[5]  = (__bf16)(b.y * scale);
    v[6]  = (__bf16)(b.z * scale); v[7]  = (__bf16)(b.w * scale);
    v[8]  = (__bf16)(c.x * scale); v[9]  = (__bf16)(c.y * scale);
    v[10] = (__bf16)(c.z * scale); v[11] = (__bf16)(c.w * scale);
    v[12] = (__bf16)(d.x * scale); v[13] = (__bf16)(d.y * scale);
    v[14] = (__bf16)(d.z * scale); v[15] = (__bf16)(d.w * scale);
    return v;
}

// Load one 16x32 (rows x K) bf16 WMMA operand tile per the CDNA5 A/B layout:
// lane half h = lane>>4, row/col = lane&15; vector elems 0..7 hold K = f0..f0+7,
// elems 8..15 hold K = f0+16..f0+23, where f0 = kbase + 8*h.
__device__ __forceinline__ v16bf load_tile_bf16(const float* __restrict__ rowp,
                                                int kbase, int half, float scale,
                                                float* ssq_accum) {
    const float* p = rowp + kbase + 8 * half;
    float4 q0 = *(const float4*)(p);
    float4 q1 = *(const float4*)(p + 4);
    float4 q2 = *(const float4*)(p + 16);
    float4 q3 = *(const float4*)(p + 20);
    if (ssq_accum) *ssq_accum += ssq4(q0) + ssq4(q1) + ssq4(q2) + ssq4(q3);
    return pack16(q0, q1, q2, q3, scale);
}

__global__ void init_ws_kernel(unsigned long long* __restrict__ ws) {
    ws[threadIdx.x] = ~0ull;
}

__global__ __launch_bounds__(256) void cluster_argmin_kernel(
    const float* __restrict__ x, const float* __restrict__ cc,
    unsigned long long* __restrict__ gmin, int n_tiles)
{
    __shared__ unsigned long long sred[NCLUS];
    if (threadIdx.x < NCLUS) sred[threadIdx.x] = ~0ull;
    __syncthreads();

    const int lane = threadIdx.x & 31;
    const int half = lane >> 4;        // which half-wave
    const int row  = lane & 15;        // A row / B col / C col held by this lane
    const int gw     = blockIdx.x * (blockDim.x >> 5) + (threadIdx.x >> 5);
    const int nwaves = gridDim.x * (blockDim.x >> 5);

    // B = (-2 * centers)^T in bf16, fully resident in VGPRs:
    // 4 cluster tiles x 4 k-steps x v16bf = 128 VGPRs. *(-2) is exact in bf16.
    v16bf Bt[4][4];
#pragma unroll
    for (int t = 0; t < 4; ++t) {
        const float* cp = cc + (size_t)(t * 16 + row) * NFEAT;
#pragma unroll
        for (int kk = 0; kk < 4; ++kk)
            Bt[t][kk] = load_tile_bf16(cp, kk * 32, half, -2.0f, nullptr);
    }

    float    vmin[4];
    unsigned imin[4];
#pragma unroll
    for (int t = 0; t < 4; ++t) { vmin[t] = __builtin_inff(); imin[t] = 0u; }

    for (int tile = gw; tile < n_tiles; tile += nwaves) {
        const int base = tile << 4;
        const float* rowp = x + (size_t)(base + row) * NFEAT;

        // Prefetch PFDIST grid-stride iterations ahead (global_prefetch_b8):
        // deep enough to cover a full HBM round trip at 2-4 waves/SIMD.
        // 2 lanes/row cover all 4 cachelines of each 512B token row.
        const int ptile = tile + PFDIST * nwaves;
        if (ptile < n_tiles) {
            const float* np = x + (size_t)((ptile << 4) + row) * NFEAT;
            __builtin_prefetch(np + 32 * half, 0, 3);
            __builtin_prefetch(np + 32 * (2 + half), 0, 3);
        }

        // Load 16 tokens x 128 features once; fp32 sum-of-squares on the fly.
        float ssq = 0.0f;
        v16bf A[4];
#pragma unroll
        for (int kk = 0; kk < 4; ++kk)
            A[kk] = load_tile_bf16(rowp, kk * 32, half, 1.0f, &ssq);

        // Full ||x_row||^2: this lane + its half-wave partner hold disjoint
        // halves of the same row's features.
        float x2row = ssq + __shfl_xor(ssq, 16);

        // Broadcast into C layout: VGPR i holds row m = 8*half + i.
        v8f c0;
#pragma unroll
        for (int i = 0; i < 8; ++i)
            c0[i] = __shfl(x2row, 8 * half + i);

        // score = ||x||^2 - 2*x.c : C initialized with x2, B carries the -2.
#pragma unroll
        for (int t = 0; t < 4; ++t) {
            v8f acc = c0;
#pragma unroll
            for (int kk = 0; kk < 4; ++kk)
                acc = __builtin_amdgcn_wmma_f32_16x16x32_bf16(
                    false, A[kk], false, Bt[t][kk], (short)0, acc, false, false);
#pragma unroll
            for (int i = 0; i < 8; ++i) {
                float s = acc[i];
                unsigned id = (unsigned)(base + 8 * half + i);
                if (s < vmin[t]) { vmin[t] = s; imin[t] = id; }
            }
        }
    }

    // Pack (sortable key | index) so u64-min == (min value, then min index).
#pragma unroll
    for (int t = 0; t < 4; ++t) {
        unsigned long long pk =
            ((unsigned long long)fkey(vmin[t]) << 32) | (unsigned long long)imin[t];
        atomicMin(&sred[t * 16 + row], pk);   // ds_min_u64
    }
    __syncthreads();
    if (threadIdx.x < NCLUS)
        atomicMin(&gmin[threadIdx.x], sred[threadIdx.x]);  // global_atomic_min_u64
}

__global__ void gather_kernel(const float* __restrict__ x,
                              const unsigned long long* __restrict__ ws,
                              float* __restrict__ out)
{
    const int k = blockIdx.x;
    const unsigned idx = (unsigned)(ws[k] & 0xffffffffull);
    out[(size_t)k * NFEAT + threadIdx.x] = x[(size_t)idx * NFEAT + threadIdx.x];
}

extern "C" void kernel_launch(void* const* d_in, const int* in_sizes, int n_in,
                              void* d_out, int out_size, void* d_ws, size_t ws_size,
                              hipStream_t stream) {
    (void)n_in; (void)out_size; (void)ws_size;
    const float* x  = (const float*)d_in[0];          // (1, N, 128) f32
    const float* cc = (const float*)d_in[1];          // (64, 128) f32
    float* out = (float*)d_out;                        // (1, 64, 128) f32
    unsigned long long* ws = (unsigned long long*)d_ws; // 64 x u64 packed (key|idx)

    const int n_tokens = in_sizes[0] / NFEAT;          // 1,000,000 (multiple of 16)
    const int n_tiles  = n_tokens / 16;                // 62,500

    init_ws_kernel<<<1, NCLUS, 0, stream>>>(ws);
    cluster_argmin_kernel<<<512, 256, 0, stream>>>(x, cc, ws, n_tiles);
    gather_kernel<<<NCLUS, NFEAT, 0, stream>>>(x, ws, out);
}
